// MaskedMultiheadSelfAttention_12841952215805
// MI455X (gfx1250) — matrix operand order repaired and checked by reading, not compiled, not run on hardware
//
#include <hip/hip_runtime.h>
#include <hip/hip_bf16.h>
#include <math.h>

// ---------------------------------------------------------------------------
// Fused causal multi-head self-attention for gfx1250 (MI455X).
// B=4, T=2048, C=1024, H=16, D=64.  All matmuls on v_wmma_f32_16x16x32_f16.
// Softmax row reductions: DPP16 row_ror rotations (max) + WMMA-vs-ones (sum).
// ---------------------------------------------------------------------------

typedef __attribute__((ext_vector_type(16))) _Float16 v16h;
typedef __attribute__((ext_vector_type(8)))  float    v8f;
typedef __attribute__((ext_vector_type(4)))  _Float16 v4h;

#define BB 4
#define TT 2048
#define CC 1024
#define HH 16
#define DD 64
#define C3 (3 * CC)
#define MROWS (BB * TT)   // 8192

union FragU { v16h v; _Float16 h[16]; uint32_t u[8]; };
union Vec8H { uint4 q; _Float16 h[8]; };

__device__ __forceinline__ v8f wmma_f16(v16h a, v16h b, v8f c) {
  // (neg_a, A, neg_b, B, c_mod, C, reuse_a, reuse_b)
  return __builtin_amdgcn_wmma_f32_16x16x32_f16(false, a, false, b, (short)0, c,
                                                false, false);
}

// A-operand 16x32 f16 fragment (ISA 7.12.2): lane row = lane&15,
// kh = (lane>>4)*8; VGPR g<4 holds K = 2g+kh,  g>=4 holds K = 16+2(g-4)+kh.
// p0 points at element (row0, k0); stride in halves; K contiguous in memory.
__device__ __forceinline__ v16h load_a16x32(const _Float16* p0, int stride) {
  const int lane = threadIdx.x & 31;
  const int row  = lane & 15;
  const int kh   = (lane >> 4) * 8;
  FragU r;
#pragma unroll
  for (int g = 0; g < 8; ++g) {
    const int d = (g < 4) ? (2 * g + kh) : (16 + 2 * (g - 4) + kh);
    r.u[g] = *(const uint32_t*)(p0 + (size_t)row * stride + d);
  }
  return r.v;
}

// B-operand 32x16 f16 fragment where memory is laid out [col][k] with k
// contiguous (stride in halves between columns). lane col = lane&15,
// ko = (lane>>4)*8; VGPR j<4 holds K = ko+2j, ko+2j+1; j>=4 holds K = 16+ko+2(j-4), +1.
__device__ __forceinline__ v16h load_b32x16_kcontig(const _Float16* p0, int stride) {
  const int lane = threadIdx.x & 31;
  const int col  = lane & 15;
  const int ko   = (lane >> 4) * 8;
  FragU r;
#pragma unroll
  for (int j = 0; j < 8; ++j)
    r.u[j] = *(const uint32_t*)(p0 + (size_t)col * stride + ((j < 4) ? (ko + 2 * j) : (16 + ko + 2 * (j - 4))));
  return r.v;
}

// 16-lane (DPP row) reduction via row_ror rotations: after ror 1,2,4,8 every
// lane of a 16-lane row holds the max over the whole row.  Pure VALU, no LDS.
template <int CTRL>
__device__ __forceinline__ float row_dpp(float x) {
  return __int_as_float(__builtin_amdgcn_update_dpp(
      0, __float_as_int(x), CTRL, 0xf, 0xf, true));
}
__device__ __forceinline__ float rmax16(float x) {
  x = fmaxf(x, row_dpp<0x121>(x));  // row_ror:1
  x = fmaxf(x, row_dpp<0x122>(x));  // row_ror:2
  x = fmaxf(x, row_dpp<0x124>(x));  // row_ror:4
  x = fmaxf(x, row_dpp<0x128>(x));  // row_ror:8
  return x;
}

// ---------------------------------------------------------------------------
// f32 -> f16 conversion (4 elements / thread)
// ---------------------------------------------------------------------------
__global__ __launch_bounds__(256) void cvt_f32_f16_kernel(
    const float* __restrict__ s, _Float16* __restrict__ d, int n4) {
  int i = blockIdx.x * blockDim.x + threadIdx.x;
  if (i < n4) {
    float4 f = ((const float4*)s)[i];
    v4h h;
    h[0] = (_Float16)f.x; h[1] = (_Float16)f.y;
    h[2] = (_Float16)f.z; h[3] = (_Float16)f.w;
    ((v4h*)d)[i] = h;
  }
}

// ---------------------------------------------------------------------------
// Tiled GEMM: C[M,N] = A[M,K] * B[K,N] + bias[N].  f16 in, f32 accum.
// Block tile 128x128, 8 waves as 4(M) x 2(N), each wave 32x64 (2x4 WMMA).
// ---------------------------------------------------------------------------
template <bool OUT_F16>
__global__ __launch_bounds__(256) void gemm_wmma_kernel(
    const _Float16* __restrict__ A, const _Float16* __restrict__ Bm,
    const float* __restrict__ bias, void* __restrict__ Cout,
    int M, int N, int K) {
  __shared__ _Float16 As[128][40];  // [row][k], 80B rows (16B-aligned chunks)
  __shared__ _Float16 Bs[128][40];  // [col][k] (transposed stage)

  const int tid   = threadIdx.x;
  const int lane  = tid & 31;
  const int wave  = tid >> 5;
  const int waveM = wave & 3;
  const int waveN = wave >> 2;
  const int blockN = blockIdx.x * 128;
  const int blockM = blockIdx.y * 128;

  v8f acc[2][4];
#pragma unroll
  for (int i = 0; i < 2; ++i)
#pragma unroll
    for (int j = 0; j < 4; ++j) acc[i][j] = {};

  for (int kb = 0; kb < K; kb += 32) {
    // Stage A tile 128x32 (uint4 = 8 halves per transfer, 2 per thread)
#pragma unroll
    for (int i = 0; i < 2; ++i) {
      const int c   = tid + i * 256;        // 0..511
      const int row = c >> 2;
      const int kc  = (c & 3) * 8;
      *(uint4*)(&As[row][kc]) =
          *(const uint4*)(A + (size_t)(blockM + row) * K + kb + kc);
    }
    // Stage B tile 32x128 transposed into [col][k] (uint4 reads, scatter write)
#pragma unroll
    for (int i = 0; i < 2; ++i) {
      const int c  = tid + i * 256;         // 0..511
      const int k  = c >> 4;                // 0..31
      const int nc = (c & 15) * 8;          // 0..120
      Vec8H t;
      t.q = *(const uint4*)(Bm + (size_t)(kb + k) * N + blockN + nc);
#pragma unroll
      for (int j = 0; j < 8; ++j) Bs[nc + j][k] = t.h[j];
    }
    // Prefetch next A tile rows (global_prefetch_b8)
    if (kb + 32 < K)
      __builtin_prefetch(A + (size_t)(blockM + (tid >> 1)) * K + kb + 32, 0, 1);
    __syncthreads();

    const v16h a0 = load_a16x32(&As[waveM * 32][0],      40);
    const v16h a1 = load_a16x32(&As[waveM * 32 + 16][0], 40);
    v16h bf[4];
#pragma unroll
    for (int j = 0; j < 4; ++j)
      bf[j] = load_b32x16_kcontig(&Bs[waveN * 64 + j * 16][0], 40);

#pragma unroll
    for (int j = 0; j < 4; ++j) {
      acc[0][j] = wmma_f16(a0, bf[j], acc[0][j]);
      acc[1][j] = wmma_f16(a1, bf[j], acc[1][j]);
    }
    __syncthreads();
  }

  // Epilogue: D layout — VGPR r holds (row = r + 8*(lane>=16), col = lane&15)
  const int hi  = lane >> 4;
  const int col = lane & 15;
#pragma unroll
  for (int i = 0; i < 2; ++i)
#pragma unroll
    for (int j = 0; j < 4; ++j)
#pragma unroll
      for (int r = 0; r < 8; ++r) {
        const int rg = blockM + waveM * 32 + i * 16 + r + 8 * hi;
        const int cg = blockN + waveN * 64 + j * 16 + col;
        const float v = acc[i][j][r] + bias[cg];
        if (OUT_F16)
          ((_Float16*)Cout)[(size_t)rg * N + cg] = (_Float16)v;
        else
          ((float*)Cout)[(size_t)rg * N + cg] = v;
      }
}

// ---------------------------------------------------------------------------
// Flash attention (causal, online softmax).  One wave = 16 query rows;
// block = 128 query rows of one (b,h).  K/V tiles staged cooperatively in
// LDS (V transposed) so all WMMA operand loads are contiguous u32 DS reads.
// Row max: DPP rotations.  Row sum: WMMA against all-ones B fragment.
// qkv: [B*T][3C] f16 (Q at col 0, K at C, V at 2C, per-head offset h*64).
// outh: [B*T][C] f16.
// ---------------------------------------------------------------------------
__global__ __launch_bounds__(256) void attn_wmma_kernel(
    const _Float16* __restrict__ qkv, _Float16* __restrict__ outh) {
  __shared__ _Float16 Ks[32][72];     // [key][d]   (d contiguous)
  __shared__ _Float16 Vt[64][40];     // [d][key]   (key contiguous)
  __shared__ _Float16 Pl[8][16][40];  // per-wave P staging (D -> A layout)

  const int tid  = threadIdx.x;
  const int lane = tid & 31;
  const int w    = tid >> 5;
  const int hh   = blockIdx.y;
  const int b    = blockIdx.z;
  const int qblock = blockIdx.x * 128;
  const int qbase  = qblock + w * 16;
  const int hi  = lane >> 4;
  const int col = lane & 15;

  // Q fragments (16 rows x 64 head-dim), pre-scaled by 1/sqrt(D) = 0.125
  const _Float16* qp = qkv + (size_t)(b * TT + qbase) * C3 + hh * DD;
  FragU qf0, qf1;
  qf0.v = load_a16x32(qp, C3);
  qf1.v = load_a16x32(qp + 32, C3);
#pragma unroll
  for (int i = 0; i < 16; ++i) {
    qf0.h[i] = (_Float16)((float)qf0.h[i] * 0.125f);
    qf1.h[i] = (_Float16)((float)qf1.h[i] * 0.125f);
  }

  // Constant all-ones B fragment: one WMMA against it yields all row sums
  // of P replicated across the 16 output columns (D layout).
  FragU onesf;
#pragma unroll
  for (int i = 0; i < 16; ++i) onesf.h[i] = (_Float16)1.0f;

  float m[8], l[8];
  v8f o[4];
#pragma unroll
  for (int r = 0; r < 8; ++r) { m[r] = -INFINITY; l[r] = 0.0f; }
#pragma unroll
  for (int nt = 0; nt < 4; ++nt) o[nt] = {};

  const _Float16* kbaseP = qkv + (size_t)(b * TT) * C3 + CC + hh * DD;
  const _Float16* vbaseP = qkv + (size_t)(b * TT) * C3 + 2 * CC + hh * DD;

  // Block-uniform tile count: keys 0 .. qblock+127.  Waves whose causal
  // frontier is below a tile see fully-masked scores -> exact no-op update.
  const int nkt = (qblock >> 5) + 4;
  const int krow = tid >> 3;        // 0..31 (key within tile)
  const int dcol = (tid & 7) * 8;   // 0..56 (d chunk)

  for (int kt = 0; kt < nkt; ++kt) {
    const int k0 = kt * 32;

    // Stage K tile straight: Ks[key][d], one uint4 per thread
    *(uint4*)(&Ks[krow][dcol]) =
        *(const uint4*)(kbaseP + (size_t)(k0 + krow) * C3 + dcol);
    // Stage V tile transposed: Vt[d][key], uint4 read + 8 scatter writes
    {
      Vec8H t;
      t.q = *(const uint4*)(vbaseP + (size_t)(k0 + krow) * C3 + dcol);
#pragma unroll
      for (int j = 0; j < 8; ++j) Vt[dcol + j][krow] = t.h[j];
    }
    __syncthreads();

    // S = Q @ K^T : two 16x16 score tiles (keys k0..+15 and k0+16..+31)
    v8f s0 = {}, s1 = {};
    {
      const v16h kb00 = load_b32x16_kcontig(&Ks[0][0],   72);
      const v16h kb01 = load_b32x16_kcontig(&Ks[0][32],  72);
      s0 = wmma_f16(qf0.v, kb00, s0);
      s0 = wmma_f16(qf1.v, kb01, s0);
      const v16h kb10 = load_b32x16_kcontig(&Ks[16][0],  72);
      const v16h kb11 = load_b32x16_kcontig(&Ks[16][32], 72);
      s1 = wmma_f16(qf0.v, kb10, s1);
      s1 = wmma_f16(qf1.v, kb11, s1);
    }

    // Causal mask + online max update (rows live in 16-lane DPP rows)
    const int kj0 = k0 + col;
    const int kj1 = k0 + 16 + col;
    float sc_[8];
#pragma unroll
    for (int r = 0; r < 8; ++r) {
      const int qi = qbase + r + 8 * hi;
      const float x0 = (kj0 <= qi) ? s0[r] : -INFINITY;
      const float x1 = (kj1 <= qi) ? s1[r] : -INFINITY;
      const float mt = rmax16(fmaxf(x0, x1));
      const float mn = fmaxf(m[r], mt);
      sc_[r] = __expf(m[r] - mn);           // exp(0)=1 when tile fully masked
      const float p0 = __expf(x0 - mn);
      const float p1 = __expf(x1 - mn);
      m[r] = mn;
#pragma unroll
      for (int nt = 0; nt < 4; ++nt) o[nt][r] *= sc_[r];
      Pl[w][r + 8 * hi][col]      = (_Float16)p0;
      Pl[w][r + 8 * hi][16 + col] = (_Float16)p1;
    }

    // Reload P in A layout, get row sums via WMMA-vs-ones, then O += P @ V
    const v16h pf = load_a16x32(&Pl[w][0][0], 40);
    v8f rs8 = {};
    rs8 = wmma_f16(pf, onesf.v, rs8);       // rs8[r] = sum_k P[row r][k]
#pragma unroll
    for (int r = 0; r < 8; ++r) l[r] = l[r] * sc_[r] + rs8[r];
#pragma unroll
    for (int nt = 0; nt < 4; ++nt) {
      const v16h vf = load_b32x16_kcontig(&Vt[nt * 16][0], 40);
      o[nt] = wmma_f16(pf, vf, o[nt]);
    }
    __syncthreads();
  }

  // Normalize and store (f16) to attention output [B*T][C]
  _Float16* op = outh + (size_t)(b * TT + qbase) * CC + hh * DD;
#pragma unroll
  for (int nt = 0; nt < 4; ++nt)
#pragma unroll
    for (int r = 0; r < 8; ++r) {
      const int row = r + 8 * hi;
      const float v = o[nt][r] / l[r];
      op[(size_t)row * CC + nt * 16 + col] = (_Float16)v;
    }
}

// ---------------------------------------------------------------------------
// Launch
// ---------------------------------------------------------------------------
extern "C" void kernel_launch(void* const* d_in, const int* in_sizes, int n_in,
                              void* d_out, int out_size, void* d_ws, size_t ws_size,
                              hipStream_t stream) {
  (void)in_sizes; (void)n_in; (void)out_size; (void)ws_size;
  const float* x     = (const float*)d_in[0];  // (B,T,C)
  const float* w_qkv = (const float*)d_in[1];  // (C,3C)
  const float* b_qkv = (const float*)d_in[2];  // (3C,)
  const float* w_out = (const float*)d_in[3];  // (C,C)
  const float* b_out = (const float*)d_in[4];  // (C,)
  float* out = (float*)d_out;                  // (B,T,C) f32

  // Workspace layout (f16), ~92.3 MB total
  _Float16* xh    = (_Float16*)d_ws;                       // 8192*1024
  _Float16* wqkvh = xh    + (size_t)MROWS * CC;            // 1024*3072
  _Float16* wouth = wqkvh + (size_t)CC * C3;               // 1024*1024
  _Float16* qkvh  = wouth + (size_t)CC * CC;               // 8192*3072
  _Float16* attnh = qkvh  + (size_t)MROWS * C3;            // 8192*1024

  // 1. downconvert inputs to f16
  cvt_f32_f16_kernel<<<(MROWS * CC / 4 + 255) / 256, 256, 0, stream>>>(
      x, xh, MROWS * CC / 4);
  cvt_f32_f16_kernel<<<(CC * C3 / 4 + 255) / 256, 256, 0, stream>>>(
      w_qkv, wqkvh, CC * C3 / 4);
  cvt_f32_f16_kernel<<<(CC * CC / 4 + 255) / 256, 256, 0, stream>>>(
      w_out, wouth, CC * CC / 4);

  // 2. QKV projection: (8192,1024) @ (1024,3072) + b_qkv -> f16
  gemm_wmma_kernel<true><<<dim3(C3 / 128, MROWS / 128), 256, 0, stream>>>(
      xh, wqkvh, b_qkv, (void*)qkvh, MROWS, C3, CC);

  // 3. causal flash attention per (qblock, head, batch)
  attn_wmma_kernel<<<dim3(TT / 128, HH, BB), 256, 0, stream>>>(qkvh, attnh);

  // 4. output projection: (8192,1024) @ (1024,1024) + b_out -> f32
  gemm_wmma_kernel<false><<<dim3(CC / 128, MROWS / 128), 256, 0, stream>>>(
      attnh, wouth, b_out, (void*)out, MROWS, CC, CC);
}
